// SpatialLocalAttention_30219389894976
// MI455X (gfx1250) — compile-verified, matrix operand order checked by reading, not run on hardware
//
#include <hip/hip_runtime.h>
#include <hip/hip_bf16.h>

typedef __bf16 bf16_t;
typedef __attribute__((ext_vector_type(16))) __bf16 v16bf;
typedef __attribute__((ext_vector_type(8)))  __bf16 v8bf;
typedef __attribute__((ext_vector_type(8)))  float  v8f;

#define NB    2
#define NL    4096
#define NK    16
#define NG    16
#define ND    256
#define NPE   64
#define NH    8
#define NDH   64
#define NC    33            // 1 + k + G
#define NCP   48            // padded to 3 WMMA M-tiles
#define DKV   320           // D + PE (contraction dim for K/V)
#define INNER 512           // H * DH
#define QSCALE 0.125f       // 64^-0.5

union ABF { v16bf full; v8bf half[2]; };

// ---------------------------------------------------------------------------
// Prologue: transpose + convert Wk / Wv (f32 [320][512]) to bf16 [512][320]
// so WMMA B-fragments are two contiguous 16B loads per lane.
// ---------------------------------------------------------------------------
__global__ void prep_weights(const float* __restrict__ Wk,
                             const float* __restrict__ Wv,
                             bf16_t* __restrict__ WkT,
                             bf16_t* __restrict__ WvT) {
  int i = blockIdx.x * blockDim.x + threadIdx.x;
  if (i >= INNER * DKV) return;
  int n  = i / DKV;
  int kk = i % DKV;
  WkT[i] = (bf16_t)Wk[(size_t)kk * INNER + n];
  WvT[i] = (bf16_t)Wv[(size_t)kk * INNER + n];
}

// ---------------------------------------------------------------------------
// Fully fused per-query attention: one workgroup (4 waves) per (b, l).
// ---------------------------------------------------------------------------
__global__ __launch_bounds__(128)
void attn_kernel(const float* __restrict__ spatial,
                 const int*   __restrict__ topk,
                 const float* __restrict__ rpe,
                 const float* __restrict__ selfrpe,
                 const float* __restrict__ dist,
                 const float* __restrict__ glat,
                 const float* __restrict__ Wq,
                 const bf16_t* __restrict__ WkT,
                 const bf16_t* __restrict__ WvT,
                 const float* __restrict__ Wo,
                 const float* __restrict__ bo,
                 const float* __restrict__ grpe,
                 const float* __restrict__ lsig,
                 const float* __restrict__ gbias,
                 float* __restrict__ out) {
  __shared__ __attribute__((aligned(16))) bf16_t ctx[NCP][DKV];   // 30 KB
  __shared__ float srow[ND];                                      //  1 KB
  __shared__ float qvec[INNER];                                   //  2 KB
  __shared__ __attribute__((aligned(16))) float kh[NCP][NDH];     // 12 KB
  __shared__ __attribute__((aligned(16))) float vh[NCP][NDH];     // 12 KB
  __shared__ float outv[INNER];                                   //  2 KB
  __shared__ float sc[NC];
  __shared__ float pr[NC];

  const int tid  = threadIdx.x;
  const int lane = tid & 31;
  const int wv   = tid >> 5;           // wave 0..3 -> N-tile within head
  const int nloc = lane & 15;          // A-row / B-col within 16-wide tile
  const int bk   = (lane < 16) ? 0 : 8;
  const int bl   = blockIdx.x;         // b*L + l
  const int b    = bl / NL;
  const size_t rowoff = (size_t)bl * ND;

  // ---- stage spatial row (f32, for Q) ----
  for (int i = tid; i < ND; i += 128) srow[i] = spatial[rowoff + i];

  // ---- build bf16 context [48 x 320]: [self | neighbors | globals] ++ rpe ----
  for (int i = tid; i < NCP * DKV; i += 128) {
    int r = i / DKV, c = i % DKV;
    float v = 0.f;
    if (r == 0) {
      v = (c < ND) ? spatial[rowoff + c]
                   : selfrpe[(size_t)bl * NPE + (c - ND)];
    } else if (r < 1 + NK) {
      int j  = r - 1;
      int nb = topk[(size_t)bl * NK + j];
      v = (c < ND) ? spatial[((size_t)b * NL + nb) * ND + c]
                   : rpe[((size_t)bl * NK + j) * NPE + (c - ND)];
    } else if (r < NC) {
      int g = r - (1 + NK);
      v = (c < ND) ? glat[((size_t)b * NG + g) * ND + c]
                   : grpe[c - ND];
    }
    ctx[r][c] = (bf16_t)v;
  }
  __syncthreads();

  // ---- Q = spatial_row @ Wq (f32 GEMV, small) ----
  for (int o = tid; o < INNER; o += 128) {
    float acc = 0.f;
    #pragma unroll 4
    for (int kk = 0; kk < ND; ++kk) acc += srow[kk] * Wq[(size_t)kk * INNER + o];
    qvec[o] = acc;
  }
  __syncthreads();

  for (int h = 0; h < NH; ++h) {
    const int ncol = h * NDH + wv * 16 + nloc;     // output column in [0,512)

    // ---- K_h / V_h = ctx[48,320] @ W[:, h*64 : h*64+64] via bf16 WMMA ----
    for (int proj = 0; proj < 2; ++proj) {
      const bf16_t* wrow = (proj ? WvT : WkT) + (size_t)ncol * DKV;
      float* dst = proj ? &vh[0][0] : &kh[0][0];
      v8f acc0 = {}, acc1 = {}, acc2 = {};
      #pragma unroll
      for (int k0 = 0; k0 < DKV / 32; ++k0) {
        const int kb = k0 * 32;
        ABF bfr, a0, a1, a2;
        bfr.half[0] = *(const v8bf*)(wrow + kb + bk);
        bfr.half[1] = *(const v8bf*)(wrow + kb + 16 + bk);
        a0.half[0] = *(const v8bf*)&ctx[nloc][kb + bk];
        a0.half[1] = *(const v8bf*)&ctx[nloc][kb + 16 + bk];
        a1.half[0] = *(const v8bf*)&ctx[16 + nloc][kb + bk];
        a1.half[1] = *(const v8bf*)&ctx[16 + nloc][kb + 16 + bk];
        a2.half[0] = *(const v8bf*)&ctx[32 + nloc][kb + bk];
        a2.half[1] = *(const v8bf*)&ctx[32 + nloc][kb + 16 + bk];
        acc0 = __builtin_amdgcn_wmma_f32_16x16x32_bf16(false, a0.full, false, bfr.full,
                                                       (short)0, acc0, false, false);
        acc1 = __builtin_amdgcn_wmma_f32_16x16x32_bf16(false, a1.full, false, bfr.full,
                                                       (short)0, acc1, false, false);
        acc2 = __builtin_amdgcn_wmma_f32_16x16x32_bf16(false, a2.full, false, bfr.full,
                                                       (short)0, acc2, false, false);
      }
      // D layout: VGPR v -> row v + (lane<16?0:8), col = lane&15 (within tile)
      const int colw = wv * 16 + nloc;
      const int rofs = (lane < 16) ? 0 : 8;
      #pragma unroll
      for (int v = 0; v < 8; ++v) {
        dst[(      v + rofs) * NDH + colw] = acc0[v];
        dst[(16 +  v + rofs) * NDH + colw] = acc1[v];
        dst[(32 +  v + rofs) * NDH + colw] = acc2[v];
      }
    }
    __syncthreads();

    // ---- scores: q_h . K_h[c] * scale + bias ----
    if (tid < NC) {
      int c = tid;
      float s = 0.f;
      #pragma unroll 8
      for (int d = 0; d < NDH; ++d) s += qvec[h * NDH + d] * kh[c][d];
      s *= QSCALE;
      float bias;
      if (c == 0) {
        bias = 0.f;
      } else if (c < 1 + NK) {
        float dd  = dist[(size_t)bl * NK + (c - 1)];
        float ssq = __expf(2.f * lsig[h]);        // sigma^2 = exp(log_sigma)^2
        bias = -(dd * dd) / (2.f * ssq);
      } else {
        bias = gbias[0];
      }
      sc[c] = s + bias;
    }
    __syncthreads();

    // ---- softmax over 33 context slots ----
    if (tid == 0) {
      float mx = sc[0];
      for (int c = 1; c < NC; ++c) mx = fmaxf(mx, sc[c]);
      float ssum = 0.f;
      for (int c = 0; c < NC; ++c) { float e = __expf(sc[c] - mx); pr[c] = e; ssum += e; }
      float inv = 1.f / ssum;
      for (int c = 0; c < NC; ++c) pr[c] *= inv;
    }
    __syncthreads();

    // ---- out_h = p . V_h ----
    if (tid < NDH) {
      int d = tid;
      float a = 0.f;
      #pragma unroll
      for (int c = 0; c < NC; ++c) a += pr[c] * vh[c][d];
      outv[h * NDH + d] = a;
    }
    __syncthreads();
  }

  // ---- final projection: y = outv @ Wo + bo ----
  for (int o = tid; o < ND; o += 128) {
    float acc = bo[o];
    #pragma unroll 4
    for (int kk = 0; kk < INNER; ++kk) acc += outv[kk] * Wo[(size_t)kk * ND + o];
    out[rowoff + o] = acc;
  }
}

extern "C" void kernel_launch(void* const* d_in, const int* in_sizes, int n_in,
                              void* d_out, int out_size, void* d_ws, size_t ws_size,
                              hipStream_t stream) {
  (void)in_sizes; (void)n_in; (void)out_size; (void)ws_size;
  const float* spatial = (const float*)d_in[0];
  const int*   topk    = (const int*)d_in[1];
  const float* rpe     = (const float*)d_in[2];
  const float* selfrpe = (const float*)d_in[3];
  const float* dist    = (const float*)d_in[4];
  const float* glat    = (const float*)d_in[5];
  const float* Wq      = (const float*)d_in[6];
  const float* Wk      = (const float*)d_in[7];
  const float* Wv      = (const float*)d_in[8];
  const float* Wo      = (const float*)d_in[9];
  const float* bo      = (const float*)d_in[10];
  const float* grpe    = (const float*)d_in[11];
  const float* lsig    = (const float*)d_in[12];
  const float* gbias   = (const float*)d_in[13];

  bf16_t* WkT = (bf16_t*)d_ws;                       // [512][320] bf16
  bf16_t* WvT = WkT + (size_t)INNER * DKV;           // [512][320] bf16

  const int nprep = INNER * DKV;
  prep_weights<<<(nprep + 255) / 256, 256, 0, stream>>>(Wk, Wv, WkT, WvT);

  attn_kernel<<<dim3(NB * NL), 128, 0, stream>>>(
      spatial, topk, rpe, selfrpe, dist, glat,
      Wq, WkT, WvT, Wo, bo, grpe, lsig, gbias, (float*)d_out);
}
